// AttentionalBiLSTM_60455959658673
// MI455X (gfx1250) — matrix operand order repaired and checked by reading, not compiled, not run on hardware
//
#include <hip/hip_runtime.h>
#include <cstdint>
#include <cstddef>

typedef __bf16 v16bf __attribute__((ext_vector_type(16)));
typedef __bf16 v8bf  __attribute__((ext_vector_type(8)));
typedef float  v8f   __attribute__((ext_vector_type(8)));

#define B_  64
#define T_  512
#define E_  256
#define C_  256
#define NB_ 8
#define QS_ 64
#define G_  1024      // 4*C gates
#define KH_ 2048      // NB_*C_ == NB_*E_

// ---- workspace layout (bytes, all offsets 256B aligned) ----
static constexpr size_t OFF_COEF  = 0;                                   // B*NB f32
static constexpr size_t OFF_BIASF = 2048;                                // B*G f32
static constexpr size_t OFF_BIASR = OFF_BIASF + (size_t)B_*G_*4;         // B*G f32
static constexpr size_t OFF_WHH   = OFF_BIASR + (size_t)B_*G_*4;         // NB*G*C bf16
static constexpr size_t OFF_WIH   = OFF_WHH  + (size_t)NB_*G_*C_*2;
static constexpr size_t OFF_WHHR  = OFF_WIH  + (size_t)NB_*G_*E_*2;
static constexpr size_t OFF_WIHR  = OFF_WHHR + (size_t)NB_*G_*C_*2;
static constexpr size_t OFF_HID   = OFF_WIHR + (size_t)NB_*G_*E_*2;      // B*T*2C f32

// ------------------------------------------------------------------
// coef[b,:] = softmax( concat(qu[usr[b]], qp[prd[b]]) @ key_w.T )
// ------------------------------------------------------------------
__global__ void coef_kernel(const float* __restrict__ qu, const float* __restrict__ qp,
                            const int* __restrict__ usr, const int* __restrict__ prd,
                            const float* __restrict__ key_w, float* __restrict__ coef) {
  int b = threadIdx.x;
  if (b >= B_) return;
  const float* u = qu + (size_t)usr[b]*QS_;
  const float* p = qp + (size_t)prd[b]*QS_;
  float lg[NB_];
  float mx = -1e30f;
  for (int n = 0; n < NB_; ++n) {
    const float* kw = key_w + n*2*QS_;
    float s = 0.f;
    for (int i = 0; i < QS_; ++i) s += u[i]*kw[i];
    for (int i = 0; i < QS_; ++i) s += p[i]*kw[QS_+i];
    lg[n] = s; mx = fmaxf(mx, s);
  }
  float den = 0.f;
  for (int n = 0; n < NB_; ++n) { lg[n] = expf(lg[n]-mx); den += lg[n]; }
  for (int n = 0; n < NB_; ++n) coef[b*NB_+n] = lg[n]/den;
}

// ------------------------------------------------------------------
// bias[b,g] = coef[b,:] @ b_in[:,g]   (both directions)
// ------------------------------------------------------------------
__global__ void bias_kernel(const float* __restrict__ coef,
                            const float* __restrict__ bin, const float* __restrict__ binr,
                            float* __restrict__ bf, float* __restrict__ br) {
  int b = blockIdx.x;
  float cf[NB_];
  for (int n = 0; n < NB_; ++n) cf[n] = coef[b*NB_+n];
  for (int g = threadIdx.x; g < G_; g += blockDim.x) {
    float s = 0.f, sr = 0.f;
    for (int n = 0; n < NB_; ++n) { s += cf[n]*bin[n*G_+g]; sr += cf[n]*binr[n*G_+g]; }
    bf[(size_t)b*G_+g] = s; br[(size_t)b*G_+g] = sr;
  }
}

// ------------------------------------------------------------------
// f32 -> bf16 basis conversion (layout preserved: (n, g, c) row-major)
// ------------------------------------------------------------------
__global__ void cvt_kernel(const float* __restrict__ in, __bf16* __restrict__ out, int n) {
  for (int i = blockIdx.x*blockDim.x + threadIdx.x; i < n; i += gridDim.x*blockDim.x)
    out[i] = (__bf16)in[i];
}

// ------------------------------------------------------------------
// Fused BiLSTM recurrence via basis-GEMM:
//   gates(16x1024) = A'(16 x 4096) x B'(4096 x 1024),  A' = coef (x) [h ; x_t]
// grid = (4 batch groups, 2 directions), block = 256 threads (8 waves)
// ------------------------------------------------------------------
__global__ __launch_bounds__(256) void recur_kernel(
    const float* __restrict__ x, const int* __restrict__ lens,
    const float* __restrict__ coef,
    const __bf16* __restrict__ whh,  const __bf16* __restrict__ wih,
    const __bf16* __restrict__ whhr, const __bf16* __restrict__ wihr,
    const float* __restrict__ bias_f, const float* __restrict__ bias_r,
    float* __restrict__ hid)
{
  extern __shared__ unsigned char smem[];
  __bf16 (*aT)[KH_] = (__bf16 (*)[KH_])smem;            // 16 x 2048 bf16 = 64KB
  float* coefS = (float*)(smem + (size_t)16*KH_*2);     // 16 x 8 f32

  const int tid  = threadIdx.x;
  const int lane = tid & 31;
  const int wv   = tid >> 5;        // wave 0..7 -> owns cells c in [wv*32, wv*32+32)
  const int hi   = lane >> 4;       // 0/1
  const int col  = lane & 15;
  const int group = blockIdx.x;     // 0..3 (16 batch samples each)
  const int dir   = blockIdx.y;     // 0=fwd, 1=rev
  const int b0 = group*16;

  const __bf16* Wh = dir ? whhr : whh;
  const __bf16* Wi = dir ? wihr : wih;
  const float* bias = dir ? bias_r : bias_f;

  if (tid < 16*NB_) coefS[tid] = coef[b0*NB_ + tid];
  __syncthreads();

  // per-lane persistent LSTM state: (m = r + 8*hi, c = wv*32 + j*16 + col)
  float h_st[2][8], c_st[2][8];
  float biasv[4][2][8];
  int lenv[8];
  for (int r = 0; r < 8; ++r) {
    const int m = r + 8*hi, bb = b0 + m;
    lenv[r] = lens[bb];
    for (int j = 0; j < 2; ++j) {
      const int c = wv*32 + j*16 + col;
      h_st[j][r] = 0.f; c_st[j][r] = 0.f;
      for (int q = 0; q < 4; ++q) biasv[q][j][r] = bias[(size_t)bb*G_ + q*256 + c];
    }
  }

  for (int t = 0; t < T_; ++t) {
    const int t_eff = dir ? (T_-1-t) : t;   // reverse scans x backwards; also the store index

    // ---- A' (h part): aT[m][n*256+c] = coef[m][n] * h[m][c]
    for (int j = 0; j < 2; ++j)
      for (int r = 0; r < 8; ++r) {
        const int m = r + 8*hi, c = wv*32 + j*16 + col;
        const float hv = h_st[j][r];
        for (int n = 0; n < NB_; ++n)
          aT[m][n*256 + c] = (__bf16)(coefS[m*NB_+n]*hv);
      }
    __syncthreads();

    v8f acc[4][2];
    {
      const v8f vz = {0.f,0.f,0.f,0.f,0.f,0.f,0.f,0.f};
      for (int q = 0; q < 4; ++q) for (int j = 0; j < 2; ++j) acc[q][j] = vz;
    }
    const __bf16* aRow = &aT[col][0];   // A row M = lane&15

    // ---- phase 1: (coef (x) h) x w_hh, K = 2048
    for (int kt = 0; kt < 64; ++kt) {
      const int k = kt*32;
      union { v16bf v; v8bf h8[2]; } af;           // A layout: lanes0-15 K{0-15}, lanes16-31 K{16-31}
      af.h8[0] = *(const v8bf*)(aRow + k + hi*16);
      af.h8[1] = *(const v8bf*)(aRow + k + hi*16 + 8);
      #pragma unroll
      for (int q = 0; q < 4; ++q)
        #pragma unroll
        for (int j = 0; j < 2; ++j) {
          const int g = q*256 + wv*32 + j*16 + col;
          const v16bf bfr = *(const v16bf*)(Wh + ((size_t)(k>>8)*G_ + g)*C_ + (k&255) + hi*16);
          acc[q][j] = __builtin_amdgcn_wmma_f32_16x16x32_bf16(
              false, af.v, false, bfr, (short)0, acc[q][j], false, false);
        }
    }
    __syncthreads();

    // ---- A' (x part): aT[m][n*256+e] = coef[m][n] * x[b][t_eff][e]
    {
      const int m = tid >> 4, e0 = (tid & 15)*16, bb = b0 + m;
      const float* xr = x + ((size_t)bb*T_ + t_eff)*E_ + e0;
      float xv[16];
      #pragma unroll
      for (int e = 0; e < 16; ++e) xv[e] = xr[e];
      for (int n = 0; n < NB_; ++n) {
        const float cfv = coefS[m*NB_+n];
        #pragma unroll
        for (int e = 0; e < 16; ++e) aT[m][n*256 + e0 + e] = (__bf16)(cfv*xv[e]);
      }
    }
    __syncthreads();

    // ---- phase 2: (coef (x) x_t) x w_ih, K = 2048, accumulate
    for (int kt = 0; kt < 64; ++kt) {
      const int k = kt*32;
      union { v16bf v; v8bf h8[2]; } af;
      af.h8[0] = *(const v8bf*)(aRow + k + hi*16);
      af.h8[1] = *(const v8bf*)(aRow + k + hi*16 + 8);
      #pragma unroll
      for (int q = 0; q < 4; ++q)
        #pragma unroll
        for (int j = 0; j < 2; ++j) {
          const int g = q*256 + wv*32 + j*16 + col;
          const v16bf bfr = *(const v16bf*)(Wi + ((size_t)(k>>8)*G_ + g)*C_ + (k&255) + hi*16);
          acc[q][j] = __builtin_amdgcn_wmma_f32_16x16x32_bf16(
              false, af.v, false, bfr, (short)0, acc[q][j], false, false);
        }
    }

    // ---- nonlinearities, state update, masked hidden store
    #pragma unroll
    for (int j = 0; j < 2; ++j)
      #pragma unroll
      for (int r = 0; r < 8; ++r) {
        const int m = r + 8*hi, bb = b0 + m;
        const int c = wv*32 + j*16 + col;
        float ig = acc[0][j][r] + biasv[0][j][r];
        float fg = acc[1][j][r] + biasv[1][j][r];
        float gg = acc[2][j][r] + biasv[2][j][r];
        float og = acc[3][j][r] + biasv[3][j][r];
        ig = 1.f/(1.f+expf(-ig));
        fg = 1.f/(1.f+expf(-fg));
        gg = tanhf(gg);
        og = 1.f/(1.f+expf(-og));
        float cy = fg*c_st[j][r] + ig*gg;
        const bool valid = dir ? (t >= T_ - lenv[r]) : (t < lenv[r]);
        const float mk = valid ? 1.f : 0.f;
        cy *= mk;
        const float hy = og * tanhf(cy) * mk;
        c_st[j][r] = cy; h_st[j][r] = hy;
        hid[((size_t)bb*T_ + t_eff)*(2*C_) + dir*C_ + c] = hy;
      }
    __syncthreads();   // protect aT before next step's rewrite
  }
}

// ------------------------------------------------------------------
// attention pooling: energy = hid.z, masked softmax over T, weighted sum
// ------------------------------------------------------------------
__global__ __launch_bounds__(256) void attn_kernel(
    const float* __restrict__ hid, const int* __restrict__ lens,
    const float* __restrict__ zw, float* __restrict__ out)
{
  __shared__ float zS[2*C_];
  __shared__ float eS[T_];
  __shared__ float red[256];
  const int b = blockIdx.x, tid = threadIdx.x;
  const int len = lens[b];
  for (int d = tid; d < 2*C_; d += 256) zS[d] = zw[d];
  __syncthreads();
  float lmx = -1e30f;
  for (int t = tid; t < T_; t += 256) {
    const float* hr = hid + ((size_t)b*T_ + t)*(2*C_);
    float s = 0.f;
    for (int d = 0; d < 2*C_; ++d) s += hr[d]*zS[d];
    if (t >= len) s = -1e30f;
    eS[t] = s; lmx = fmaxf(lmx, s);
  }
  red[tid] = lmx; __syncthreads();
  for (int off = 128; off > 0; off >>= 1) { if (tid < off) red[tid] = fmaxf(red[tid], red[tid+off]); __syncthreads(); }
  const float mx = red[0]; __syncthreads();
  float lsum = 0.f;
  for (int t = tid; t < T_; t += 256) { float e = expf(eS[t]-mx); eS[t] = e; lsum += e; }
  red[tid] = lsum; __syncthreads();
  for (int off = 128; off > 0; off >>= 1) { if (tid < off) red[tid] += red[tid+off]; __syncthreads(); }
  const float inv = 1.f/red[0];
  __syncthreads();
  for (int d = tid; d < 2*C_; d += 256) {
    float s = 0.f;
    for (int t = 0; t < T_; ++t) s += eS[t]*hid[((size_t)b*T_+t)*(2*C_)+d];
    out[(size_t)b*(2*C_)+d] = s*inv;
  }
}

// ------------------------------------------------------------------
extern "C" void kernel_launch(void* const* d_in, const int* in_sizes, int n_in,
                              void* d_out, int out_size, void* d_ws, size_t ws_size,
                              hipStream_t stream) {
  (void)in_sizes; (void)n_in; (void)out_size; (void)ws_size;
  const float* x      = (const float*)d_in[0];
  const int*   lens   = (const int*)d_in[1];
  const int*   usr    = (const int*)d_in[2];
  const int*   prd    = (const int*)d_in[3];
  const float* qu     = (const float*)d_in[4];
  const float* qp     = (const float*)d_in[5];
  const float* key_w  = (const float*)d_in[6];
  const float* w_ih   = (const float*)d_in[7];
  const float* w_hh   = (const float*)d_in[8];
  const float* bin    = (const float*)d_in[9];
  const float* w_ih_r = (const float*)d_in[10];
  const float* w_hh_r = (const float*)d_in[11];
  const float* bin_r  = (const float*)d_in[12];
  const float* zw     = (const float*)d_in[13];
  float* out = (float*)d_out;
  char* ws = (char*)d_ws;

  float*  coef  = (float*)(ws + OFF_COEF);
  float*  biasf = (float*)(ws + OFF_BIASF);
  float*  biasr = (float*)(ws + OFF_BIASR);
  __bf16* whh   = (__bf16*)(ws + OFF_WHH);
  __bf16* wih   = (__bf16*)(ws + OFF_WIH);
  __bf16* whhr  = (__bf16*)(ws + OFF_WHHR);
  __bf16* wihr  = (__bf16*)(ws + OFF_WIHR);
  float*  hid   = (float*)(ws + OFF_HID);

  coef_kernel<<<1, 64, 0, stream>>>(qu, qp, usr, prd, key_w, coef);
  bias_kernel<<<B_, 256, 0, stream>>>(coef, bin, bin_r, biasf, biasr);
  const int nW = NB_*G_*C_;
  cvt_kernel<<<512, 256, 0, stream>>>(w_hh,   whh,  nW);
  cvt_kernel<<<512, 256, 0, stream>>>(w_ih,   wih,  nW);
  cvt_kernel<<<512, 256, 0, stream>>>(w_hh_r, whhr, nW);
  cvt_kernel<<<512, 256, 0, stream>>>(w_ih_r, wihr, nW);

  const size_t shmem = (size_t)16*KH_*2 + (size_t)16*NB_*4;  // 64KB A' tile + coef (gfx1250 WGP has 320KB LDS)
  recur_kernel<<<dim3(4,2), 256, shmem, stream>>>(x, lens, coef, whh, wih, whhr, wihr, biasf, biasr, hid);
  attn_kernel<<<B_, 256, 0, stream>>>(hid, lens, zw, out);
}
